// AttnBlockpp_44564580663321
// MI455X (gfx1250) — compile-verified, matrix operand order checked by reading, not run on hardware
//
#include <hip/hip_runtime.h>

// ---------------------------------------------------------------------------
// Types for CDNA5 WMMA (wave32)
// ---------------------------------------------------------------------------
typedef __bf16 bf16;
typedef __attribute__((ext_vector_type(16))) __bf16 v16bf;
typedef __attribute__((ext_vector_type(8)))  float  v8f;
typedef int v4i __attribute__((ext_vector_type(4)));

typedef __attribute__((address_space(1))) v4i gas_v4i;   // global
typedef __attribute__((address_space(3))) v4i las_v4i;   // LDS

union FragBF {
    uint4  u[2];   // 32 bytes
    v16bf  v;
};

// Problem constants
#define BB   16
#define CC   512
#define HW   1024          // 32*32
#define NG   32            // groups
#define CPG  16            // channels per group
#define MTOT (BB * HW)     // 16384 rows for folded-batch GEMMs

// Async global->LDS copy support (gfx1250 ASYNCcnt path)
#if defined(__has_builtin)
#if __has_builtin(__builtin_amdgcn_global_load_async_to_lds_b128)
#define HAVE_ASYNC_LDS 1
#endif
#endif

__device__ __forceinline__ void wait_asynccnt0() {
#if defined(__has_builtin) && __has_builtin(__builtin_amdgcn_s_wait_asynccnt)
    __builtin_amdgcn_s_wait_asynccnt(0);
#else
    asm volatile("s_wait_asynccnt 0" ::: "memory");
#endif
}

// ---------------------------------------------------------------------------
// Load a 16x32 bf16 A/B fragment from a row-major [rows x ld] matrix.
// Lane layout (CDNA5 ISA 7.12.2, 16-bit A 16x32):
//   lanes 0-15 : row M=lane,     VGPR0-3 = K 0..7,  VGPR4-7 = K 16..23
//   lanes 16-31: row M=lane-16,  VGPR0-3 = K 8..15, VGPR4-7 = K 24..31
// ---------------------------------------------------------------------------
__device__ __forceinline__ v16bf load_frag(const bf16* __restrict__ base, int ld) {
    const int lane = threadIdx.x & 31;
    const int row  = lane & 15;
    const int kh   = (lane >> 4) << 3;          // 0 or 8
    const bf16* p  = base + (long)row * ld + kh;
    FragBF f;
    f.u[0] = *reinterpret_cast<const uint4*>(p);
    f.u[1] = *reinterpret_cast<const uint4*>(p + 16);
    return f.v;
}

__device__ __forceinline__ v8f wmma_bf16(v16bf a, v16bf b, v8f c) {
    return __builtin_amdgcn_wmma_f32_16x16x32_bf16(
        /*neg_a=*/false, a, /*neg_b=*/false, b,
        /*c_mod=*/(short)0, c, /*reuse_a=*/false, /*reuse_b=*/false);
}

// ---------------------------------------------------------------------------
// 1) Weight transpose + f32 -> bf16 cast:  wt[d*C + c] = w[c*C + d]
// ---------------------------------------------------------------------------
__global__ __launch_bounds__(256) void wconv_kernel(const float* __restrict__ w,
                                                    bf16* __restrict__ wt) {
    int idx = blockIdx.x * 256 + threadIdx.x;          // over C*C
    if (idx >= CC * CC) return;
    int c = idx / CC;
    int d = idx - c * CC;
    wt[(long)d * CC + c] = (bf16)w[idx];
}

// ---------------------------------------------------------------------------
// 2) GroupNorm: x[B,C,HW] f32  ->  hn[B,HW,C] bf16
//    one block per (b, g); 16 channels * 1024 positions = 16384 elements
// ---------------------------------------------------------------------------
__global__ __launch_bounds__(256) void groupnorm_kernel(const float* __restrict__ x,
                                                        const float* __restrict__ gamma,
                                                        const float* __restrict__ beta,
                                                        bf16* __restrict__ hn) {
    __shared__ float rs[256], rss[256];
    const int b = blockIdx.x / NG;
    const int g = blockIdx.x - b * NG;
    const float* base = x + ((long)b * CC + (long)g * CPG) * HW;

    float s = 0.f, ss = 0.f;
    for (int i = threadIdx.x; i < CPG * HW; i += 256) {
        float v = base[i];
        s += v; ss += v * v;
    }
    rs[threadIdx.x] = s; rss[threadIdx.x] = ss;
    __syncthreads();
    for (int stp = 128; stp > 0; stp >>= 1) {
        if (threadIdx.x < stp) {
            rs[threadIdx.x]  += rs[threadIdx.x + stp];
            rss[threadIdx.x] += rss[threadIdx.x + stp];
        }
        __syncthreads();
    }
    const float inv_n = 1.0f / (CPG * HW);
    const float mu  = rs[0] * inv_n;
    const float var = rss[0] * inv_n - mu * mu;
    const float rsd = rsqrtf(var + 1e-6f);

    for (int i = threadIdx.x; i < CPG * HW; i += 256) {
        int cc = i / HW;                // 0..15
        int hw = i - cc * HW;
        int c  = g * CPG + cc;
        float v = (base[i] - mu) * rsd * gamma[c] + beta[c];
        hn[((long)b * HW + hw) * CC + c] = (bf16)v;
    }
}

// ---------------------------------------------------------------------------
// 3) Generic batched WMMA GEMM:
//      out[z, m, n] = sum_k A[z, m, k] * Bop[z, n, k]  (+ bias[n])
//    A   : bf16 [Z, M, K]    row-major, stride a_bs per z
//    Bop : bf16 [Z, Nout, K] row-major (i.e. B transposed), stride b_bs per z
//    MODE 0: bf16 out, [Z*M, Nout] row-major (ldo = Nout)
//    MODE 1: bf16 out, transposed per image: out[b, n, pos] (pos = row & 1023)
//    MODE 2: f32  out, transposed per image, + residual res (same layout)
//    Block: 256 threads = 8 waves as 2(M) x 4(N); wave tile 32 x 64
//    (each B fragment reused across 2 A rows -> 8 WMMA per 12 b128 loads).
//    Block tile: 64(M) x 256(N).
// ---------------------------------------------------------------------------
template <int MODE>
__global__ __launch_bounds__(256) void gemm_wmma(const bf16* __restrict__ A, int lda, long a_bs,
                                                 const bf16* __restrict__ Bop, int ldb, long b_bs,
                                                 const float* __restrict__ bias,
                                                 void* __restrict__ out,
                                                 const float* __restrict__ res,
                                                 int M, int K, int Nout) {
    const int z    = blockIdx.z;
    const bf16* a  = A   + (long)z * a_bs;
    const bf16* bp = Bop + (long)z * b_bs;

    const int tid  = threadIdx.x;
    const int wave = tid >> 5;
    const int wm   = wave & 1;
    const int wn   = wave >> 1;
    const int m0   = blockIdx.x * 64  + wm * 32;
    const int n0   = blockIdx.y * 256 + wn * 64;

    v8f acc[2][4] = {};

    const bf16* arow0 = a + (long)m0 * lda;
    const bf16* arow1 = a + (long)(m0 + 16) * lda;
    for (int k = 0; k < K; k += 32) {
        // prefetch next k-chunk of A (emits global_prefetch_b8 on gfx1250)
        if (k + 32 < K) {
            __builtin_prefetch((const void*)(arow0 + k + 32), 0, 0);
            __builtin_prefetch((const void*)(arow1 + k + 32), 0, 0);
        }
        v16bf af0 = load_frag(arow0 + k, lda);
        v16bf af1 = load_frag(arow1 + k, lda);
#pragma unroll
        for (int t = 0; t < 4; ++t) {
            v16bf bfrag = load_frag(bp + (long)(n0 + t * 16) * ldb + k, ldb);
            acc[0][t] = wmma_bf16(af0, bfrag, acc[0][t]);
            acc[1][t] = wmma_bf16(af1, bfrag, acc[1][t]);
        }
    }

    // D layout: VGPR r -> (m = r + 8*(lane>>4), n = lane&15)
    const int lane = tid & 31;
    const int nloc = lane & 15;
    const int mhi  = (lane >> 4) << 3;

#pragma unroll
    for (int mi = 0; mi < 2; ++mi) {
#pragma unroll
        for (int t = 0; t < 4; ++t) {
            const int n  = n0 + t * 16 + nloc;
            const float bv = bias ? bias[n] : 0.0f;
#pragma unroll
            for (int r = 0; r < 8; ++r) {
                const int  m    = m0 + mi * 16 + mhi + r;
                const long grow = (long)z * M + m;
                const float val = acc[mi][t][r] + bv;
                if (MODE == 0) {
                    ((bf16*)out)[grow * Nout + n] = (bf16)val;
                } else if (MODE == 1) {
                    const long bi  = grow >> 10;        // /1024 (HW)
                    const long pos = grow & 1023;
                    ((bf16*)out)[bi * (long)CC * HW + (long)n * HW + pos] = (bf16)val;
                } else {
                    const long bi  = grow >> 10;
                    const long pos = grow & 1023;
                    const long idx = bi * (long)CC * HW + (long)n * HW + pos;
                    ((float*)out)[idx] = res[idx] + val;
                }
            }
        }
    }
}

// ---------------------------------------------------------------------------
// 4) Fused scores + softmax:  P[b, i, :] = softmax( Q[b,i,:] . K[b,j,:] / sqrt(C) )
//    One block (256 threads, 8 waves) handles 16 query rows of one image.
//    Q tile staged into LDS via async global->LDS (ASYNCcnt) when available.
//    Each wave handles 8 key tiles as 2 groups of 4 (Q fragment reused 4x).
//    Dynamic LDS: q-tile 16x512 bf16 (16KB) + scores 16x1024 f32 (64KB) + red.
// ---------------------------------------------------------------------------
__global__ __launch_bounds__(256) void attn_scores_softmax(const bf16* __restrict__ q,
                                                           const bf16* __restrict__ kk,
                                                           bf16* __restrict__ p) {
    extern __shared__ char smem[];
    bf16*  qs  = (bf16*)smem;                         // 16*512 bf16
    float* sc  = (float*)(smem + 16 * CC * 2);        // 16*1024 f32
    float* red = sc + 16 * HW;                        // 256 f32
    float* rst = red + 256;                           // 16 f32

    const int bidx = blockIdx.x >> 6;                 // image
    const int qt   = blockIdx.x & 63;                 // 16-query tile
    const bf16* qbase = q  + ((long)bidx * HW + (long)qt * 16) * CC;
    const bf16* kbase = kk + (long)bidx * HW * CC;
    const int tid = threadIdx.x;

    // stage Q tile into LDS (1024 x 16B)
    {
        const uint4* src = (const uint4*)qbase;
        uint4*       dst = (uint4*)qs;
#if defined(HAVE_ASYNC_LDS)
        for (int i = tid; i < (16 * CC * 2) / 16; i += 256) {
            __builtin_amdgcn_global_load_async_to_lds_b128(
                (gas_v4i*)(void*)(src + i),
                (las_v4i*)(void*)(dst + i),
                /*offset=*/0, /*cpol=*/0);
        }
        wait_asynccnt0();
#else
        for (int i = tid; i < (16 * CC * 2) / 16; i += 256) dst[i] = src[i];
#endif
    }
    __syncthreads();

    const int wave = tid >> 5;
    const float scale = 0.044194173824159216f;        // 1/sqrt(512)
    const int lane = tid & 31;
    const int nloc = lane & 15;
    const int mhi  = (lane >> 4) << 3;

#pragma unroll
    for (int grp = 0; grp < 2; ++grp) {
        const int jt0 = wave * 8 + grp * 4;           // 4 consecutive 16-key tiles
        v8f acc[4] = {};
        const bf16* kt = kbase + (long)jt0 * 16 * CC;
        for (int ks = 0; ks < CC; ks += 32) {
            v16bf af = load_frag(qs + ks, CC);        // ds_read path, reused 4x
#pragma unroll
            for (int t = 0; t < 4; ++t) {
                v16bf bf_ = load_frag(kt + (long)t * 16 * CC + ks, CC);
                acc[t] = wmma_bf16(af, bf_, acc[t]);
            }
        }
#pragma unroll
        for (int t = 0; t < 4; ++t)
#pragma unroll
            for (int r = 0; r < 8; ++r)
                sc[(mhi + r) * HW + (jt0 + t) * 16 + nloc] = acc[t][r] * scale;
    }
    __syncthreads();

    // block softmax over 16 rows of 1024; 16 threads per row
    const int r    = tid >> 4;
    const int cpar = tid & 15;
    float mx = -1e30f;
    for (int j = cpar; j < HW; j += 16) mx = fmaxf(mx, sc[r * HW + j]);
    red[tid] = mx;
    __syncthreads();
    if (cpar == 0) {
        float m2 = red[r * 16];
        for (int i = 1; i < 16; ++i) m2 = fmaxf(m2, red[r * 16 + i]);
        rst[r] = m2;
    }
    __syncthreads();
    const float rowm = rst[r];
    float s = 0.f;
    for (int j = cpar; j < HW; j += 16) {
        float e = __expf(sc[r * HW + j] - rowm);
        sc[r * HW + j] = e;
        s += e;
    }
    red[tid] = s;
    __syncthreads();
    if (cpar == 0) {
        float s2 = 0.f;
        for (int i = 0; i < 16; ++i) s2 += red[r * 16 + i];
        rst[r] = 1.0f / s2;
    }
    __syncthreads();
    const float inv = rst[r];
    bf16* prow = p + ((long)bidx * HW + (long)qt * 16 + r) * HW;
    for (int j = cpar; j < HW; j += 16)
        prow[j] = (bf16)(sc[r * HW + j] * inv);
}

// ---------------------------------------------------------------------------
// Launch
// ---------------------------------------------------------------------------
extern "C" void kernel_launch(void* const* d_in, const int* in_sizes, int n_in,
                              void* d_out, int out_size, void* d_ws, size_t ws_size,
                              hipStream_t stream) {
    const float* x     = (const float*)d_in[0];
    const float* gamma = (const float*)d_in[1];
    const float* beta  = (const float*)d_in[2];
    const float* W0 = (const float*)d_in[3];  const float* b0 = (const float*)d_in[4];
    const float* W1 = (const float*)d_in[5];  const float* b1 = (const float*)d_in[6];
    const float* W2 = (const float*)d_in[7];  const float* b2 = (const float*)d_in[8];
    const float* W3 = (const float*)d_in[9];  const float* b3 = (const float*)d_in[10];

    // workspace carve-up
    size_t off = 0;
    auto carve = [&](size_t bytes) -> void* {
        void* pp = (char*)d_ws + off;
        off += (bytes + 255) & ~(size_t)255;
        return pp;
    };
    bf16* hn  = (bf16*)carve((size_t)MTOT * CC * 2);
    bf16* wt0 = (bf16*)carve((size_t)CC * CC * 2);
    bf16* wt1 = (bf16*)carve((size_t)CC * CC * 2);
    bf16* wt2 = (bf16*)carve((size_t)CC * CC * 2);
    bf16* wt3 = (bf16*)carve((size_t)CC * CC * 2);
    bf16* qb  = (bf16*)carve((size_t)MTOT * CC * 2);
    bf16* kb  = (bf16*)carve((size_t)MTOT * CC * 2);
    bf16* vt  = (bf16*)carve((size_t)BB * CC * HW * 2);   // [B, C, HW]
    bf16* P   = (bf16*)carve((size_t)BB * HW * HW * 2);   // [B, HW, HW]
    bf16* O   = (bf16*)carve((size_t)MTOT * CC * 2);

    const int wblocks = (CC * CC + 255) / 256;
    wconv_kernel<<<wblocks, 256, 0, stream>>>(W0, wt0);
    wconv_kernel<<<wblocks, 256, 0, stream>>>(W1, wt1);
    wconv_kernel<<<wblocks, 256, 0, stream>>>(W2, wt2);
    wconv_kernel<<<wblocks, 256, 0, stream>>>(W3, wt3);

    groupnorm_kernel<<<BB * NG, 256, 0, stream>>>(x, gamma, beta, hn);

    // q = hn @ W0 + b0 ; k = hn @ W1 + b1 ; vt = (hn @ W2 + b2)^T per image
    dim3 gproj(MTOT / 64, CC / 256, 1);
    gemm_wmma<0><<<gproj, 256, 0, stream>>>(hn, CC, 0, wt0, CC, 0, b0, qb, nullptr, MTOT, CC, CC);
    gemm_wmma<0><<<gproj, 256, 0, stream>>>(hn, CC, 0, wt1, CC, 0, b1, kb, nullptr, MTOT, CC, CC);
    gemm_wmma<1><<<gproj, 256, 0, stream>>>(hn, CC, 0, wt2, CC, 0, b2, vt, nullptr, MTOT, CC, CC);

    // P = softmax(q k^T / sqrt(C)) per image
    const size_t smem = (size_t)16 * CC * 2 + (size_t)16 * HW * 4 + 256 * 4 + 16 * 4;
    attn_scores_softmax<<<BB * (HW / 16), 256, smem, stream>>>(qb, kb, P);

    // O = P @ V  (Bop = vt, rows = output channels, K = keys)
    dim3 gpv(HW / 64, CC / 256, BB);
    gemm_wmma<0><<<gpv, 256, 0, stream>>>(P, HW, (long)HW * HW,
                                          vt, HW, (long)CC * HW,
                                          nullptr, O, nullptr, HW, HW, CC);

    // out = x + (O @ W3 + b3), f32, [B, C, HW]
    gemm_wmma<2><<<gproj, 256, 0, stream>>>(O, CC, 0, wt3, CC, 0, b3, d_out, x, MTOT, CC, CC);
}
